// GCN_41240275976787
// MI455X (gfx1250) — matrix operand order repaired and checked by reading, not compiled, not run on hardware
//
#include <hip/hip_runtime.h>
#include <hip/hip_bf16.h>
#include <math.h>

typedef __attribute__((ext_vector_type(16))) _Float16 v16h;
typedef __attribute__((ext_vector_type(8)))  float    v8f;
typedef __attribute__((ext_vector_type(4)))  unsigned int v4u;

// ---------------------------------------------------------------------------
// Degree / normalization (computed once, shared by all 4 layers)
// ---------------------------------------------------------------------------
__global__ void k_fill1(float* __restrict__ p, int n) {
    int i = blockIdx.x * blockDim.x + threadIdx.x;
    if (i < n) p[i] = 1.0f;                      // self-loop contributes 1
}

__global__ void k_deg(const int* __restrict__ dst, int e, float* __restrict__ deg) {
    int i = blockIdx.x * blockDim.x + threadIdx.x;
    if (i < e) atomicAdd(&deg[dst[i]], 1.0f);
}

__global__ void k_rsqrt(float* __restrict__ p, int n) {
    int i = blockIdx.x * blockDim.x + threadIdx.x;
    if (i < n) p[i] = rsqrtf(p[i]);              // deg >= 1 always (self loop)
}

// ---------------------------------------------------------------------------
// fp32 -> padded fp16 convert, optional fused ReLU (feeds the WMMA GEMM)
// out is [n, 1<<kpadShift] fp16, zero-padded beyond K
// ---------------------------------------------------------------------------
__global__ void k_cvt(const float* __restrict__ in, _Float16* __restrict__ out,
                      int n, int K, int kpadShift, int relu) {
    int idx = blockIdx.x * blockDim.x + threadIdx.x;
    int total = n << kpadShift;
    if (idx >= total) return;
    int i = idx >> kpadShift;
    int k = idx & ((1 << kpadShift) - 1);
    float v = 0.0f;
    if (k < K) {
        v = in[i * K + k];
        if (relu) v = fmaxf(v, 0.0f);
    }
    out[idx] = (_Float16)v;
}

// ---------------------------------------------------------------------------
// Pre-pack W[K, NOUT] (fp32, row-major) into WMMA B-fragment layout (f16):
//   Bp[((kt*NT + nt)*32 + lane)*16 + v] = W[kt*32 + (v<8 ? 8*(lane>>4) + v : 16 + 8*(lane>>4) + v-8)][nt*16 + (lane&15)]
// All divergence / guards / converts happen once here, not in the GEMM loop.
// ---------------------------------------------------------------------------
template<int KPAD, int NOUT, int NT>
__global__ void k_pack(const float* __restrict__ W, _Float16* __restrict__ Bp, int K) {
    const int KT = KPAD / 32;
    int t = blockIdx.x * blockDim.x + threadIdx.x;   // one thread per (frag, lane)
    if (t >= KT * NT * 32) return;
    int frag = t >> 5;
    int lane = t & 31;
    int kt = frag / NT, nt = frag % NT;
    int half = lane >> 4, mr = lane & 15;
    int n = nt * 16 + mr;
    for (int v = 0; v < 16; ++v) {
        int k = kt * 32 + (v < 8 ? 8 * half + v : 16 + 8 * half + (v - 8));
        float w = (k < K && n < NOUT) ? W[k * NOUT + n] : 0.0f;
        Bp[(size_t)t * 16 + v] = (_Float16)w;
    }
}

// ---------------------------------------------------------------------------
// WMMA GEMM: hW[N, NOUT] = xh[N, KPAD](f16) @ W (pre-packed B frags, f16)
// B fragments live in registers for the whole kernel; waves grid-stride over
// 16-node tiles. Inner body: 2x b128 A-loads + NT wmma per K-tile, then
// 8*NT immediate-offset stores off ONE per-lane base pointer.
// ---------------------------------------------------------------------------
template<int KPAD, int NOUT, int NT>
__global__ __launch_bounds__(256)
void k_gemm(const _Float16* __restrict__ xh, const _Float16* __restrict__ Bp,
            float* __restrict__ out, int ntiles) {
    const int KT = KPAD / 32;
    int lane = threadIdx.x & 31;
    int half = lane >> 4;                        // wave32: lanes 0-15 / 16-31
    int mr   = lane & 15;

    // Load all B fragments for this layer into registers (coalesced b128 x2 each)
    union BF { v16h h; v4u q[2]; };
    BF b[KT * NT];
#pragma unroll
    for (int i = 0; i < KT * NT; ++i) {
        const _Float16* bp = Bp + ((size_t)i * 32 + lane) * 16;
        b[i].q[0] = *(const v4u*)(bp);
        b[i].q[1] = *(const v4u*)(bp + 8);
    }

    int nwaves = (int)((gridDim.x * blockDim.x) >> 5);
    int tile0  = (int)((blockIdx.x * blockDim.x + threadIdx.x) >> 5);

    // Per-lane bases; advance by the grid stride each iteration so the loop
    // body has no index arithmetic beyond two pointer bumps.
    const _Float16* arow = xh + (size_t)(tile0 * 16 + mr) * KPAD + 8 * half;
    float* orow = out + (size_t)(tile0 * 16 + 8 * half) * NOUT + mr;
    const size_t astep = (size_t)nwaves * 16 * KPAD;
    const size_t ostep = (size_t)nwaves * 16 * NOUT;

    for (int tile = tile0; tile < ntiles; tile += nwaves) {
        v8f c[NT] = {};
#pragma unroll
        for (int kt = 0; kt < KT; ++kt) {
            // A fragment (16x32 f16): lane needs K in [8h,8h+8) and [16+8h,16+8h+8)
            union { v16h h; v4u q[2]; } a;
            a.q[0] = *(const v4u*)(arow + kt * 32);
            a.q[1] = *(const v4u*)(arow + kt * 32 + 16);
#pragma unroll
            for (int nt = 0; nt < NT; ++nt) {
                c[nt] = __builtin_amdgcn_wmma_f32_16x16x32_f16(
                            false, a.h, false, b[kt * NT + nt].h,
                            (short)0, c[nt], false, false);
            }
        }
        // D layout: VGPR r, lane l -> (m = r + 8*half, n = (l&15) + 16*nt)
        // All 8*NT stores share one base; offsets are compile-time constants.
#pragma unroll
        for (int nt = 0; nt < NT; ++nt) {
            if (nt * 16 + mr < NOUT) {
#pragma unroll
                for (int r = 0; r < 8; ++r)
                    orow[r * NOUT + nt * 16] = c[nt][r];
            }
        }
        arow += astep;
        orow += ostep;
    }
}

// ---------------------------------------------------------------------------
// out[i,f] = b[f] + hW[i,f] * dinv[i]^2     (bias + self-loop term)
// ---------------------------------------------------------------------------
__global__ void k_init(float* __restrict__ out, const float* __restrict__ hW,
                       const float* __restrict__ b, const float* __restrict__ dinv,
                       int n, int noutShift) {
    int idx = blockIdx.x * blockDim.x + threadIdx.x;
    int total = n << noutShift;
    if (idx >= total) return;
    int i = idx >> noutShift;
    int f = idx & ((1 << noutShift) - 1);
    float di = dinv[i];
    out[idx] = b[f] + hW[idx] * di * di;
}

// ---------------------------------------------------------------------------
// Edge scatter: out[dst,f] += hW[src,f] * dinv[src]*dinv[dst]
// feature dim innermost -> coalesced gather + cacheline-clustered atomics
// ---------------------------------------------------------------------------
__global__ void k_scatter(float* __restrict__ out, const float* __restrict__ hW,
                          const int* __restrict__ src, const int* __restrict__ dst,
                          const float* __restrict__ dinv, int e, int noutShift) {
    int idx = blockIdx.x * blockDim.x + threadIdx.x;
    int total = e << noutShift;
    if (idx >= total) return;
    int eid = idx >> noutShift;
    int f   = idx & ((1 << noutShift) - 1);
    int s = src[eid];
    int d = dst[eid];
    float v = hW[((size_t)s << noutShift) + f] * dinv[s] * dinv[d];
    atomicAdd(&out[((size_t)d << noutShift) + f], v);
}

// ---------------------------------------------------------------------------
// log_softmax over 2 classes
// ---------------------------------------------------------------------------
__global__ void k_lsm2(const float* __restrict__ in, float* __restrict__ out, int n) {
    int i = blockIdx.x * blockDim.x + threadIdx.x;
    if (i >= n) return;
    float a = in[2 * i], b = in[2 * i + 1];
    float m = fmaxf(a, b);
    float lse = m + logf(expf(a - m) + expf(b - m));
    out[2 * i]     = a - lse;
    out[2 * i + 1] = b - lse;
}

// ---------------------------------------------------------------------------
static inline size_t alignup(size_t x) { return (x + 255) & ~(size_t)255; }
static inline unsigned cdiv(long long a, int b) { return (unsigned)((a + b - 1) / b); }

extern "C" void kernel_launch(void* const* d_in, const int* in_sizes, int n_in,
                              void* d_out, int out_size, void* d_ws, size_t ws_size,
                              hipStream_t stream) {
    const float* x  = (const float*)d_in[0];
    const int*   ei = (const int*)d_in[1];
    const float* W1 = (const float*)d_in[2]; const float* b1 = (const float*)d_in[3];
    const float* W2 = (const float*)d_in[4]; const float* b2 = (const float*)d_in[5];
    const float* W3 = (const float*)d_in[6]; const float* b3 = (const float*)d_in[7];
    const float* W4 = (const float*)d_in[8]; const float* b4 = (const float*)d_in[9];

    const int N = in_sizes[0] / 18;
    const int E = in_sizes[1] / 2;
    const int* src = ei;
    const int* dst = ei + E;

    char* ws = (char*)d_ws;
    float*    dinv = (float*)ws;      ws += alignup((size_t)N * 4);
    _Float16* Bp   = (_Float16*)ws;   ws += alignup((size_t)8 * 32 * 16 * 2); // room for 8 frags; max KT*NT is 4
    _Float16* xh   = (_Float16*)ws;   ws += alignup((size_t)N * 64 * 2);
    float*    P0   = (float*)ws;      ws += alignup((size_t)N * 64 * 4);
    float*    P1   = (float*)ws;

    const int T = 256;
    const int ntiles = (N + 15) / 16;            // 6250 exactly for N=100000
    const unsigned gemmBlocks = 256;             // 2048 waves grid-striding over tiles

    // --- symmetric normalization (shared across layers) ---
    k_fill1<<<cdiv(N, T), T, 0, stream>>>(dinv, N);
    k_deg  <<<cdiv(E, T), T, 0, stream>>>(dst, E, dinv);
    k_rsqrt<<<cdiv(N, T), T, 0, stream>>>(dinv, N);

    // --- layer 1: 18 -> 64 ---
    k_cvt<<<cdiv((long long)N * 32, T), T, 0, stream>>>(x, xh, N, 18, 5, 0);
    k_pack<32, 64, 4><<<1, 128, 0, stream>>>(W1, Bp, 18);
    k_gemm<32, 64, 4><<<gemmBlocks, T, 0, stream>>>(xh, Bp, P1, ntiles);
    k_init   <<<cdiv((long long)N * 64, T), T, 0, stream>>>(P0, P1, b1, dinv, N, 6);
    k_scatter<<<cdiv((long long)E * 64, T), T, 0, stream>>>(P0, P1, src, dst, dinv, E, 6);

    // --- layer 2: 64 -> 32 (ReLU fused into convert) ---
    k_cvt<<<cdiv((long long)N * 64, T), T, 0, stream>>>(P0, xh, N, 64, 6, 1);
    k_pack<64, 32, 2><<<1, 128, 0, stream>>>(W2, Bp, 64);
    k_gemm<64, 32, 2><<<gemmBlocks, T, 0, stream>>>(xh, Bp, P1, ntiles);
    k_init   <<<cdiv((long long)N * 32, T), T, 0, stream>>>(P0, P1, b2, dinv, N, 5);
    k_scatter<<<cdiv((long long)E * 32, T), T, 0, stream>>>(P0, P1, src, dst, dinv, E, 5);

    // --- layer 3: 32 -> 16 ---
    k_cvt<<<cdiv((long long)N * 32, T), T, 0, stream>>>(P0, xh, N, 32, 5, 1);
    k_pack<32, 16, 1><<<1, 128, 0, stream>>>(W3, Bp, 32);
    k_gemm<32, 16, 1><<<gemmBlocks, T, 0, stream>>>(xh, Bp, P1, ntiles);
    k_init   <<<cdiv((long long)N * 16, T), T, 0, stream>>>(P0, P1, b3, dinv, N, 4);
    k_scatter<<<cdiv((long long)E * 16, T), T, 0, stream>>>(P0, P1, src, dst, dinv, E, 4);

    // --- layer 4: 16 -> 2 (pad K to 32, mask output columns) ---
    k_cvt<<<cdiv((long long)N * 32, T), T, 0, stream>>>(P0, xh, N, 16, 5, 1);
    k_pack<32, 2, 1><<<1, 128, 0, stream>>>(W4, Bp, 16);
    k_gemm<32, 2, 1><<<gemmBlocks, T, 0, stream>>>(xh, Bp, P1, ntiles);
    k_init   <<<cdiv((long long)N * 2, T), T, 0, stream>>>(P0, P1, b4, dinv, N, 1);
    k_scatter<<<cdiv((long long)E * 2, T), T, 0, stream>>>(P0, P1, src, dst, dinv, E, 1);

    // --- log_softmax -> d_out ---
    k_lsm2<<<cdiv(N, T), T, 0, stream>>>(P0, (float*)d_out, N);
}
